// TacticAIGNN_45621142618521
// MI455X (gfx1250) — compile-verified, matrix operand order checked by reading, not run on hardware
//
#include <hip/hip_runtime.h>

// ---------------- problem constants (match reference) ----------------
#define NN 20000
#define EE 320000
#define GG 1000
#define F_NODE 12
#define F_EDGE 4
#define HID 64
#define HEADS 4
#define NCLS 4
#define NT 4          // col-tiles (16 cols each) per wave in the WMMA GEMM
#define KT_MAX 8      // max K tiles (Kpad=256 -> 8)

typedef __attribute__((ext_vector_type(16))) __bf16 v16bf;
typedef __attribute__((ext_vector_type(8)))  __bf16 v8bf;
typedef __attribute__((ext_vector_type(8)))  float  v8f;
typedef __attribute__((ext_vector_type(4)))  unsigned v4u;
typedef __attribute__((ext_vector_type(8)))  int      v8i;
typedef __attribute__((ext_vector_type(4)))  int      v4i;

// ---------------- helpers ----------------
__device__ __forceinline__ unsigned mapf(float f) {
    unsigned u = __float_as_uint(f);
    return (u & 0x80000000u) ? ~u : (u | 0x80000000u);
}
__device__ __forceinline__ float unmapf(unsigned u) {
    return (u & 0x80000000u) ? __uint_as_float(u & 0x7FFFFFFFu)
                             : __uint_as_float(~u);
}

__global__ void zero_kernel(float* p, size_t n) {
    size_t i = (size_t)blockIdx.x * blockDim.x + threadIdx.x;
    if (i < n) p[i] = 0.0f;
}

// fp32 activations [M,Kact] -> bf16 [M,Kpad], zero-padded tail.
__global__ void cvt_act_kernel(const float* __restrict__ A, __bf16* __restrict__ Ab,
                               int M, int Kact, int Kpad) {
    long tid = (long)blockIdx.x * blockDim.x + threadIdx.x;
    if (tid >= (long)M * Kpad) return;
    int row = (int)(tid / Kpad);
    int k   = (int)(tid % Kpad);
    Ab[tid] = (k < Kact) ? (__bf16)A[(size_t)row * Kact + k] : (__bf16)0.0f;
}

// fp32 weight [Kact,Nout] -> bf16 packed in WMMA B-fragment order:
//   Wp[((nt*(Kpad/32)+kt)*32 + lane)*16 + e], e -> k = kt*32 + (e/8)*16 + (lane/16)*8 + e%8,
//   col = nt*16 + lane%16.  Fragment read in GEMM = one contiguous 32B per lane.
__global__ void pack_weight_kernel(const float* __restrict__ W, __bf16* __restrict__ Wp,
                                   int Kact, int Kpad, int Nout) {
    int tid = blockIdx.x * blockDim.x + threadIdx.x;
    if (tid >= Kpad * Nout) return;
    int e    = tid & 15;
    int lane = (tid >> 4) & 31;
    int kTiles = Kpad >> 5;
    int kt   = (tid >> 9) % kTiles;
    int nt   = (tid >> 9) / kTiles;
    int col  = nt * 16 + (lane & 15);
    int k    = kt * 32 + ((e >> 3) << 4) + ((lane >> 4) << 3) + (e & 7);
    Wp[tid] = (k < Kact) ? (__bf16)W[(size_t)k * Nout + col] : (__bf16)0.0f;
}

// ---------------- WMMA GEMM: C[M,Nout] = Ab[M,Kpad](bf16) @ W + bias ----------------
// Block = 8 waves, each computing 16 rows x 64 cols.  All waves share the same packed-B
// slab (NT*kTiles*1KB <= 32KB), so it is DMA'd into LDS once per block with the Tensor
// Data Mover, then fragments are read with ds_load_b128.  A stays in global (b128 loads).
__global__ __launch_bounds__(256)
void gemm_wmma_kernel(const __bf16* __restrict__ Ab, const __bf16* __restrict__ Wp,
                      const float* __restrict__ bias, float* __restrict__ Cmat,
                      int M, int Kpad, int Nout) {
    __shared__ __align__(16) __bf16 sB[NT * KT_MAX * 512];   // 32 KB max

    int wave   = threadIdx.x >> 5;
    int lane   = threadIdx.x & 31;
    int halfId = lane >> 4;
    int r16    = lane & 15;
    int mTile  = blockIdx.y * 8 + wave;
    int row    = mTile * 16 + r16;
    int rowC   = row < M ? row : (M - 1);
    int kTiles = Kpad >> 5;

    // ---- stage packed B for this block's col range into LDS via TDM ----
    const __bf16* gB = Wp + ((size_t)(blockIdx.x * NT) * kTiles) * 512;
    unsigned nbytes  = (unsigned)(NT * kTiles * 512) * 2u;       // <= 32768
    unsigned nelem8  = nbytes >> 3;                              // 8-byte elements, <= 4096
    if (threadIdx.x < 32) {                                      // wave 0 issues the DMA
        unsigned long long ga = (unsigned long long)(uintptr_t)gB;
        unsigned loff = (unsigned)(uintptr_t)(&sB[0]);           // LDS window offset
        v4u g0 = { 1u,                                           // count=1 (valid D#)
                   loff,                                         // lds_addr
                   (unsigned)(ga & 0xFFFFFFFFu),                 // global_addr[31:0]
                   (unsigned)((ga >> 32) & 0x1FFFFFFu) | (2u << 30) };  // [56:32] | type=2
        v8i g1 = { (int)(3u << 16),                              // data_size=8B, mask=0
                   (int)((nelem8 & 0xFFFFu) << 16),              // tensor_dim0[15:0]
                   (int)((nelem8 >> 16) | (1u << 16)),           // tensor_dim0[31:16] | tdim1=1
                   (int)((nelem8 & 0xFFFFu) << 16),              // tile_dim0
                   0,                                            // tile_dim1/2 = 0 (1-D)
                   (int)nelem8,                                  // tensor_dim0_stride[31:0]
                   0, 0 };
        v4i gz4 = { 0, 0, 0, 0 };
        v8i gz8 = { 0, 0, 0, 0, 0, 0, 0, 0 };
        __builtin_amdgcn_tensor_load_to_lds(g0, g1, gz4, gz4, gz8, 0);
        __builtin_amdgcn_s_wait_tensorcnt(0);
    }
    __syncthreads();

    const __bf16* Arow  = Ab + (size_t)rowC * Kpad + halfId * 8;
    const __bf16* sBl   = sB + (size_t)lane * 16;

    v8f acc[NT] = {};
    for (int kt = 0; kt < kTiles; ++kt) {
        int kbase = kt << 5;
        union { v16bf v; v8bf h[2]; } ua;
        ua.h[0] = *(const v8bf*)(Arow + kbase);
        ua.h[1] = *(const v8bf*)(Arow + kbase + 16);
        int kbn = (kt + 1 < kTiles ? kbase + 32 : kbase);
        __builtin_prefetch(Arow + kbn, 0, 3);
#pragma unroll
        for (int nt = 0; nt < NT; ++nt) {
            v16bf b = *(const v16bf*)(sBl + ((size_t)nt * kTiles + kt) * 512);
            acc[nt] = __builtin_amdgcn_wmma_f32_16x16x32_bf16(false, ua.v, false, b,
                                                              (short)0, acc[nt], false, false);
        }
    }
#pragma unroll
    for (int nt = 0; nt < NT; ++nt) {
        int col = (blockIdx.x * NT + nt) * 16 + r16;
        float bv = bias[col];
#pragma unroll
        for (int r = 0; r < 8; ++r) {
            int orow = mTile * 16 + halfId * 8 + r;
            if (orow < M) Cmat[(size_t)orow * Nout + col] = acc[nt][r] + bv;
        }
    }
}

// ---------------- edge logits + segment max ----------------
// C = H*64 channels. 8 channels/thread (two float4 loads), C/8 threads per edge.
__global__ void edge_logits_kernel(const float* __restrict__ xl, const float* __restrict__ xr,
                                   const float* __restrict__ ea, const float* __restrict__ We,
                                   const float* __restrict__ att,
                                   const int* __restrict__ src, const int* __restrict__ dst,
                                   float* __restrict__ logits, unsigned* __restrict__ maxes,
                                   int C, int H) {
    extern __shared__ float sm[];
    float* sWe  = sm;           // 4*C
    float* sAtt = sm + 4 * C;   // C
    for (int i = threadIdx.x; i < 4 * C; i += blockDim.x) sWe[i] = We[i];
    for (int i = threadIdx.x; i < C; i += blockDim.x)     sAtt[i] = att[i];
    __syncthreads();
    int lpe = C >> 3;
    long tid = (long)blockIdx.x * blockDim.x + threadIdx.x;
    long e = tid / lpe;
    int  sub = (int)(tid % lpe);
    if (e >= EE) return;
    int s = src[e], d = dst[e];
    float4 ev = *(const float4*)(ea + e * 4);
    int c0 = sub * 8;
    float4 xa0 = *(const float4*)(xl + (size_t)s * C + c0);
    float4 xa1 = *(const float4*)(xl + (size_t)s * C + c0 + 4);
    float4 xb0 = *(const float4*)(xr + (size_t)d * C + c0);
    float4 xb1 = *(const float4*)(xr + (size_t)d * C + c0 + 4);
    float xs[8] = {xa0.x + xb0.x, xa0.y + xb0.y, xa0.z + xb0.z, xa0.w + xb0.w,
                   xa1.x + xb1.x, xa1.y + xb1.y, xa1.z + xb1.z, xa1.w + xb1.w};
    float a = 0.0f;
#pragma unroll
    for (int i = 0; i < 8; ++i) {
        int cc = c0 + i;
        float z = xs[i] + ev.x * sWe[cc] + ev.y * sWe[C + cc]
                + ev.z * sWe[2 * C + cc] + ev.w * sWe[3 * C + cc];
        z = z > 0.0f ? z : 0.2f * z;                 // leaky_relu 0.2
        a += z * sAtt[cc];
    }
    a += __shfl_xor(a, 1, 8);
    a += __shfl_xor(a, 2, 8);
    a += __shfl_xor(a, 4, 8);
    if ((sub & 7) == 0) {
        int h = c0 >> 6;
        logits[e * H + h] = a;
        atomicMax(&maxes[(size_t)d * H + h], mapf(a));
    }
}

// ---------------- exp(logit - max) and segment-sum denominator ----------------
__global__ void edge_exp_kernel(const float* __restrict__ logits, const int* __restrict__ dst,
                                const unsigned* __restrict__ maxes,
                                float* __restrict__ exb, float* __restrict__ denom, int H) {
    long tid = (long)blockIdx.x * blockDim.x + threadIdx.x;
    if (tid >= (long)EE * H) return;
    long e = tid / H;
    int  h = (int)(tid % H);
    int  d = dst[e];
    float m = unmapf(maxes[(size_t)d * H + h]);
    float ex = expf(logits[tid] - m);
    exb[tid] = ex;
    atomicAdd(&denom[(size_t)d * H + h], ex);
}

// ---------------- message scatter: out[dst] += xl[src] * alpha ----------------
__global__ void scatter_msg_kernel(const float* __restrict__ xl, const int* __restrict__ src,
                                   const int* __restrict__ dst, const float* __restrict__ exb,
                                   const float* __restrict__ denom, float* __restrict__ acc,
                                   int C, int H) {
    int lpe = C >> 3;
    long tid = (long)blockIdx.x * blockDim.x + threadIdx.x;
    long e = tid / lpe;
    int  sub = (int)(tid % lpe);
    if (e >= EE) return;
    int s = src[e], d = dst[e];
    int c0 = sub * 8;
    int h = c0 >> 6;
    float alpha = exb[e * H + h] / (denom[(size_t)d * H + h] + 1e-16f);
    float4 xa0 = *(const float4*)(xl + (size_t)s * C + c0);
    float4 xa1 = *(const float4*)(xl + (size_t)s * C + c0 + 4);
    float xs[8] = {xa0.x, xa0.y, xa0.z, xa0.w, xa1.x, xa1.y, xa1.z, xa1.w};
    float* accd = acc + (size_t)d * C + c0;
#pragma unroll
    for (int i = 0; i < 8; ++i) atomicAdd(&accd[i], xs[i] * alpha);
}

// ---------------- bias + optional ELU ----------------
__global__ void finish_kernel(const float* __restrict__ acc, const float* __restrict__ bias,
                              float* __restrict__ out, size_t total, int C, int do_elu) {
    size_t tid = (size_t)blockIdx.x * blockDim.x + threadIdx.x;
    if (tid >= total) return;
    int c = (int)(tid % C);
    float v = acc[tid] + bias[c];
    if (do_elu) v = v > 0.0f ? v : (expf(v) - 1.0f);
    out[tid] = v;
}

// ---------------- global mean pool (sum + count) ----------------
__global__ void pool_kernel(const float* __restrict__ h, const int* __restrict__ batch,
                            float* __restrict__ pooled, float* __restrict__ counts) {
    long tid = (long)blockIdx.x * blockDim.x + threadIdx.x;
    if (tid >= (long)NN * HID) return;
    int n = (int)(tid >> 6);
    int c = (int)(tid & 63);
    int g = batch[n];
    atomicAdd(&pooled[(size_t)g * HID + c], h[tid]);
    if (c == 0) atomicAdd(&counts[g], 1.0f);
}

// ---------------- per-graph MLP head ----------------
__global__ void mlp_kernel(const float* __restrict__ pooled, const float* __restrict__ counts,
                           const float* __restrict__ W1, const float* __restrict__ b1,
                           const float* __restrict__ W2, const float* __restrict__ b2,
                           const float* __restrict__ W3, const float* __restrict__ b3,
                           float* __restrict__ out) {
    __shared__ float gv[HID];
    __shared__ float a1[32];
    __shared__ float a2[16];
    int g = blockIdx.x;
    int t = threadIdx.x;
    float cnt = fmaxf(counts[g], 1.0f);
    gv[t] = pooled[(size_t)g * HID + t] / cnt;
    __syncthreads();
    if (t < 32) {
        float s = b1[t];
        for (int c = 0; c < HID; ++c) s += gv[c] * W1[c * 32 + t];
        a1[t] = fmaxf(s, 0.0f);
    }
    __syncthreads();
    if (t < 16) {
        float s = b2[t];
        for (int c = 0; c < 32; ++c) s += a1[c] * W2[c * 16 + t];
        a2[t] = fmaxf(s, 0.0f);
    }
    __syncthreads();
    if (t < NCLS) {
        float s = b3[t];
        for (int c = 0; c < 16; ++c) s += a2[c] * W3[c * NCLS + t];
        out[(size_t)g * NCLS + t] = s;
    }
}

// ---------------- host-side orchestration ----------------
static inline int blocks_for(long n, int bs) { return (int)((n + bs - 1) / bs); }

extern "C" void kernel_launch(void* const* d_in, const int* in_sizes, int n_in,
                              void* d_out, int out_size, void* d_ws, size_t ws_size,
                              hipStream_t stream) {
    (void)in_sizes; (void)n_in; (void)out_size; (void)ws_size;
    const float* x     = (const float*)d_in[0];
    const int*   ei    = (const int*)d_in[1];
    const float* ea    = (const float*)d_in[2];
    const int*   batch = (const int*)d_in[3];
    const int* src = ei;
    const int* dst = ei + EE;

    const float* Wl[3]   = {(const float*)d_in[4],  (const float*)d_in[11], (const float*)d_in[18]};
    const float* bl[3]   = {(const float*)d_in[5],  (const float*)d_in[12], (const float*)d_in[19]};
    const float* Wr[3]   = {(const float*)d_in[6],  (const float*)d_in[13], (const float*)d_in[20]};
    const float* br[3]   = {(const float*)d_in[7],  (const float*)d_in[14], (const float*)d_in[21]};
    const float* We[3]   = {(const float*)d_in[8],  (const float*)d_in[15], (const float*)d_in[22]};
    const float* att[3]  = {(const float*)d_in[9],  (const float*)d_in[16], (const float*)d_in[23]};
    const float* bias[3] = {(const float*)d_in[10], (const float*)d_in[17], (const float*)d_in[24]};
    const float* mW1 = (const float*)d_in[25]; const float* mb1 = (const float*)d_in[26];
    const float* mW2 = (const float*)d_in[27]; const float* mb2 = (const float*)d_in[28];
    const float* mW3 = (const float*)d_in[29]; const float* mb3 = (const float*)d_in[30];

    // ---- carve workspace (float granularity; all chunks 32B-aligned) ----
    float* ws = (float*)d_ws;
    size_t o = 0;
    float* hA     = ws + o; o += (size_t)NN * 256;
    float* hB     = ws + o; o += (size_t)NN * 256;
    float* xl     = ws + o; o += (size_t)NN * 256;
    float* xr     = ws + o; o += (size_t)NN * 256;
    float* acc    = ws + o; o += (size_t)NN * 256;
    float* logits = ws + o; o += (size_t)EE * HEADS;
    float* exb    = ws + o; o += (size_t)EE * HEADS;
    unsigned* maxes = (unsigned*)(ws + o); o += (size_t)NN * HEADS;
    float* denom  = ws + o; o += (size_t)NN * HEADS;
    float* pooled = ws + o; o += (size_t)GG * HID;
    float* counts = ws + o; o += (size_t)GG + 64;
    __bf16* Ab  = (__bf16*)(ws + o); o += (size_t)NN * 256 / 2;   // bf16 activations, padded K
    __bf16* Wlp = (__bf16*)(ws + o); o += (256 * 256) / 2;        // packed bf16 weights
    __bf16* Wrp = (__bf16*)(ws + o); o += (256 * 256) / 2;

    const int BS = 256;

    auto run_layer = [&](const float* A, int Kact, int Kpad, int H,
                         const float* Wl_, const float* bl_, const float* Wr_, const float* br_,
                         const float* We_, const float* att_, const float* bias_,
                         float* outbuf, int do_elu) {
        int C = H * HID;
        // pack weights into WMMA fragment order; convert activations to padded bf16
        int nW = Kpad * C;
        pack_weight_kernel<<<blocks_for(nW, BS), BS, 0, stream>>>(Wl_, Wlp, Kact, Kpad, C);
        pack_weight_kernel<<<blocks_for(nW, BS), BS, 0, stream>>>(Wr_, Wrp, Kact, Kpad, C);
        cvt_act_kernel<<<blocks_for((long)NN * Kpad, BS), BS, 0, stream>>>(A, Ab, NN, Kact, Kpad);

        dim3 gg(C / (16 * NT), (NN + 127) / 128);
        gemm_wmma_kernel<<<gg, BS, 0, stream>>>(Ab, Wlp, bl_, xl, NN, Kpad, C);
        gemm_wmma_kernel<<<gg, BS, 0, stream>>>(Ab, Wrp, br_, xr, NN, Kpad, C);

        zero_kernel<<<blocks_for((long)NN * H, BS), BS, 0, stream>>>((float*)maxes, (size_t)NN * H);
        zero_kernel<<<blocks_for((long)NN * H, BS), BS, 0, stream>>>(denom, (size_t)NN * H);
        zero_kernel<<<blocks_for((long)NN * C, BS), BS, 0, stream>>>(acc, (size_t)NN * C);

        long nt = (long)EE * (C / 8);
        size_t smem = (size_t)(5 * C) * sizeof(float);
        edge_logits_kernel<<<blocks_for(nt, BS), BS, smem, stream>>>(
            xl, xr, ea, We_, att_, src, dst, logits, maxes, C, H);
        edge_exp_kernel<<<blocks_for((long)EE * H, BS), BS, 0, stream>>>(
            logits, dst, maxes, exb, denom, H);
        scatter_msg_kernel<<<blocks_for(nt, BS), BS, 0, stream>>>(
            xl, src, dst, exb, denom, acc, C, H);
        finish_kernel<<<blocks_for((long)NN * C, BS), BS, 0, stream>>>(
            acc, bias_, outbuf, (size_t)NN * C, C, do_elu);
    };

    // layer 1: x[N,12] -> hA[N,256] (ELU)
    run_layer(x,  F_NODE, 32,  HEADS, Wl[0], bl[0], Wr[0], br[0], We[0], att[0], bias[0], hA, 1);
    // layer 2: hA[N,256] -> hB[N,256] (ELU)
    run_layer(hA, 256,    256, HEADS, Wl[1], bl[1], Wr[1], br[1], We[1], att[1], bias[1], hB, 1);
    // layer 3: hB[N,256] -> hA[N,64] (no activation), H=1
    run_layer(hB, 256,    256, 1,     Wl[2], bl[2], Wr[2], br[2], We[2], att[2], bias[2], hA, 0);

    // global mean pool + MLP head
    zero_kernel<<<blocks_for((long)GG * HID, BS), BS, 0, stream>>>(pooled, (size_t)GG * HID);
    zero_kernel<<<blocks_for((long)GG, BS), BS, 0, stream>>>(counts, (size_t)GG);
    pool_kernel<<<blocks_for((long)NN * HID, BS), BS, 0, stream>>>(hA, batch, pooled, counts);
    mlp_kernel<<<GG, HID, 0, stream>>>(pooled, counts, mW1, mb1, mW2, mb2, mW3, mb3,
                                       (float*)d_out);
}
